// MultiHeadAttention_41154376631032
// MI455X (gfx1250) — matrix operand order repaired and checked by reading, not compiled, not run on hardware
//
#include <hip/hip_runtime.h>
#include <hip/hip_bf16.h>

// ---------------------------------------------------------------------------
// MHA forward for MI455X (gfx1250, wave32, WMMA, async-to-LDS staging).
// Pipeline: cvt(f32->bf16) -> GEMM(Q) -> GEMM(K) -> GEMM(V, transposed store)
//           -> flash-attention (WMMA QK^T + online softmax + WMMA PV)
//           -> GEMM(out-proj, f32 out)
// Tiles staged with GLOBAL_LOAD_ASYNC_TO_LDS_B128, double-buffered against
// s_wait_asynccnt (exactly 4 async issues per thread per stage). Fragment
// loads hoisted ahead of the WMMA blocks so ds_loads batch under one wait and
// the 8 WMMAs issue back-to-back.
// ---------------------------------------------------------------------------

typedef __attribute__((ext_vector_type(16))) __bf16 v16bf;
typedef __attribute__((ext_vector_type(8)))  float  v8f;
typedef __attribute__((ext_vector_type(4)))  unsigned short us4;

#define IND   1024
#define BATCH 4
#define SEQ   2048
#define NH    16
#define DH    64
#define MTOT  (BATCH * SEQ)   // 8192

__device__ __forceinline__ unsigned short f2bf(float f) {
  union { float f; unsigned int u; } c; c.f = f;
  unsigned int u = c.u;
  u += 0x7fffu + ((u >> 16) & 1u);   // round-to-nearest-even
  return (unsigned short)(u >> 16);
}

union FragAB { v16bf v; uint4 q[2]; };
union FragC  { v8f   v; float f[8]; };

// A-matrix 16x32 bf16 fragment (ISA 7.12.2): lane holds row M=lane&15;
// VGPR0..3 = K[8*(lane>=16) .. +7], VGPR4..7 = K[16+8*(lane>=16) .. +7].
__device__ __forceinline__ v16bf load_frag_a(const unsigned short* base, int ld, int lane) {
  FragAB fr;
  const int row = lane & 15;
  const int kh  = (lane >> 4) << 3;              // 0 or 8
  const unsigned short* p = base + row * ld + kh;
  fr.q[0] = *(const uint4*)(p);
  fr.q[1] = *(const uint4*)(p + 16);
  return fr.v;
}

// B-matrix 32x16 bf16 fragment: lane holds column N=lane&15;
// VGPR0..3 = K[8*(lane>=16) .. +7], VGPR4..7 = K[16+8*(lane>=16) .. +7].
__device__ __forceinline__ v16bf load_frag_b(const unsigned short* base, int ld, int lane) {
  FragAB fr;
  const int n  = lane & 15;
  const int kh = (lane >> 4) << 3;               // 0 or 8
  const unsigned short* p = base + n * ld + kh;
  fr.q[0] = *(const uint4*)(p);
  fr.q[1] = *(const uint4*)(p + 16);
  return fr.v;
}

// ---- CDNA5 async memory->LDS copy (16B per lane), tracked by ASYNCcnt ------
__device__ __forceinline__ void async_ld16(unsigned short* lds_dst,
                                           const unsigned short* gsrc) {
  unsigned lds_off = (unsigned)(uintptr_t)lds_dst;          // low 32 bits = LDS offset
  unsigned long long ga = (unsigned long long)(uintptr_t)gsrc;
  asm volatile("global_load_async_to_lds_b128 %0, %1, off"
               :: "v"(lds_off), "v"(ga) : "memory");
}
__device__ __forceinline__ void wait_async4() {
  asm volatile("s_wait_asynccnt 0x4" ::: "memory");
}
__device__ __forceinline__ void wait_async0() {
  asm volatile("s_wait_asynccnt 0x0" ::: "memory");
}

// ---------------------------------------------------------------------------
// Kernel 1: f32 -> bf16 (n multiple of 4)
// ---------------------------------------------------------------------------
__global__ void cvt_kernel(const float* __restrict__ in,
                           unsigned short* __restrict__ out, int n) {
  int i = (blockIdx.x * 256 + threadIdx.x) * 4;
  if (i < n) {
    float4 f = *(const float4*)(in + i);
    us4 o;
    o.x = f2bf(f.x); o.y = f2bf(f.y); o.z = f2bf(f.z); o.w = f2bf(f.w);
    *(us4*)(out + i) = o;
  }
}

// ---------------------------------------------------------------------------
// Kernel 2/4: C[M=8192, N=1024] = A(bf16) @ W(bf16)^T + bias   (NT GEMM)
//   mode 0: scatter bf16 to [B,H,S,D]   (Q, K)
//   mode 1: scatter bf16 to [B,H,D,S]   (V transposed for PV B-operand)
//   mode 2: f32 to [M,N]                (final output)
// 256 thr / 8 waves, tile 128x128, K-slab 32, double-buffered async staging.
// ---------------------------------------------------------------------------
__global__ __launch_bounds__(256)
void gemm_qkvo(const unsigned short* __restrict__ A,
               const unsigned short* __restrict__ W,
               const float* __restrict__ bias,
               unsigned short* __restrict__ outb,
               float* __restrict__ outf,
               int mode) {
  __shared__ __align__(16) unsigned short As[2][128 * 40];  // 32 K + 8 pad
  __shared__ __align__(16) unsigned short Bs[2][128 * 40];

  const int tid  = threadIdx.x;
  const int lane = tid & 31;
  const int wave = tid >> 5;
  const int wm   = wave >> 1;               // 0..3  (M)
  const int wn   = wave & 1;                // 0..1  (N)
  const int tm   = blockIdx.y * 128;
  const int tn   = blockIdx.x * 128;

  // stage one 128x32 K-slab of A and W: straight-line, exactly 4 async/thread
  const int sr = tid >> 2;                  // 0..63
  const int sc = (tid & 3) << 3;            // 0,8,16,24
  auto stage = [&](int buf, int k0) {
    async_ld16(&As[buf][sr * 40 + sc],        &A[(size_t)(tm + sr)      * IND + k0 + sc]);
    async_ld16(&Bs[buf][sr * 40 + sc],        &W[(size_t)(tn + sr)      * IND + k0 + sc]);
    async_ld16(&As[buf][(sr + 64) * 40 + sc], &A[(size_t)(tm + sr + 64) * IND + k0 + sc]);
    async_ld16(&Bs[buf][(sr + 64) * 40 + sc], &W[(size_t)(tn + sr + 64) * IND + k0 + sc]);
  };

  FragC acc[2][4];
  for (int i = 0; i < 2; ++i)
    for (int j = 0; j < 4; ++j)
      for (int e = 0; e < 8; ++e) acc[i][j].f[e] = 0.f;

  const int NIT = IND / 32;                 // 32 K-steps
  stage(0, 0);
  for (int it = 0; it < NIT; ++it) {
    const int buf = it & 1;
    if (it + 1 < NIT) { stage(buf ^ 1, (it + 1) * 32); wait_async4(); }
    else              { wait_async0(); }
    __syncthreads();                        // slab `it` resident for all waves

    // hoist all fragment loads, then issue the 8 WMMAs back-to-back
    v16bf af[2], bfr[4];
    af[0] = load_frag_a(&As[buf][(wm * 32 + 0)  * 40], 40, lane);
    af[1] = load_frag_a(&As[buf][(wm * 32 + 16) * 40], 40, lane);
#pragma unroll
    for (int nt = 0; nt < 4; ++nt)
      bfr[nt] = load_frag_b(&Bs[buf][(wn * 64 + nt * 16) * 40], 40, lane);
#pragma unroll
    for (int nt = 0; nt < 4; ++nt)
#pragma unroll
      for (int mt = 0; mt < 2; ++mt)
        acc[mt][nt].v = __builtin_amdgcn_wmma_f32_16x16x32_bf16(
            false, af[mt], false, bfr[nt], (short)0, acc[mt][nt].v, false, false);

    __syncthreads();                        // done reading; next stage may overwrite
  }

  const int hi = lane >> 4, col = lane & 15;   // C layout: row = r + 8*hi, N = col
#pragma unroll
  for (int mt = 0; mt < 2; ++mt)
#pragma unroll
    for (int nt = 0; nt < 4; ++nt)
#pragma unroll
      for (int r = 0; r < 8; ++r) {
        int m = tm + wm * 32 + mt * 16 + r + 8 * hi;
        int n = tn + wn * 64 + nt * 16 + col;
        float v = acc[mt][nt].f[r] + bias[n];
        if (mode == 0) {
          int b = m >> 11, s = m & (SEQ - 1);
          int h = n >> 6,  d = n & (DH - 1);
          outb[(((size_t)(b * NH + h)) * SEQ + s) * DH + d] = f2bf(v);
        } else if (mode == 1) {
          int b = m >> 11, s = m & (SEQ - 1);
          int h = n >> 6,  d = n & (DH - 1);
          outb[(((size_t)(b * NH + h)) * DH + d) * SEQ + s] = f2bf(v);
        } else {
          outf[(size_t)m * IND + n] = v;
        }
      }
}

// ---------------------------------------------------------------------------
// Kernel 3: flash attention per (b,h). Block = 128 query rows, 8 waves x 16.
// Q A-fragments loaded straight from global (per-wave, once). K/V streamed in
// 64-wide tiles, double-buffered via async-to-LDS. Online softmax with
// xor-shuffle row reductions (16-lane halves share a C-fragment row, wave32).
// ---------------------------------------------------------------------------
__global__ __launch_bounds__(256)
void attn_kernel(const unsigned short* __restrict__ Q,   // [B,H,S,D] bf16
                 const unsigned short* __restrict__ K,   // [B,H,S,D] bf16
                 const unsigned short* __restrict__ Vt,  // [B,H,D,S] bf16
                 unsigned short* __restrict__ vw) {      // [B*S, H*D] bf16
  __shared__ __align__(16) unsigned short Ks[2][64 * 72];   // z x d
  __shared__ __align__(16) unsigned short Vs[2][64 * 72];   // d x z
  __shared__ __align__(16) unsigned short Ps[128 * 72];     // wave-private slabs

  const int tid  = threadIdx.x;
  const int lane = tid & 31;
  const int wave = tid >> 5;
  const int q0   = blockIdx.x * 128;
  const int bh   = blockIdx.y;              // b*NH + h
  const size_t qkBase = (size_t)bh * SEQ * DH;
  const size_t vtBase = (size_t)bh * DH * SEQ;

  // stage one 64-row K tile + 64x64 Vt tile: straight-line, 4 async/thread
  const int sr = tid >> 3;                  // 0..31
  const int sc = (tid & 7) << 3;            // 0..56
  auto stageKV = [&](int buf, int z0) {
    async_ld16(&Ks[buf][sr * 72 + sc],        &K[qkBase + (size_t)(z0 + sr)      * DH + sc]);
    async_ld16(&Vs[buf][sr * 72 + sc],        &Vt[vtBase + (size_t)sr        * SEQ + z0 + sc]);
    async_ld16(&Ks[buf][(sr + 32) * 72 + sc], &K[qkBase + (size_t)(z0 + sr + 32) * DH + sc]);
    async_ld16(&Vs[buf][(sr + 32) * 72 + sc], &Vt[vtBase + (size_t)(sr + 32) * SEQ + z0 + sc]);
  };

  // Q fragments for this wave's 16 rows, straight from global (row-major, ld=DH)
  const unsigned short* qbase = Q + qkBase + (size_t)(q0 + wave * 16) * DH;
  v16bf qa0 = load_frag_a(qbase + 0,  DH, lane);
  v16bf qa1 = load_frag_a(qbase + 32, DH, lane);

  FragC o[4];
  for (int j = 0; j < 4; ++j)
    for (int e = 0; e < 8; ++e) o[j].f[e] = 0.f;
  float rowmax[8], rowsum[8];
  for (int r = 0; r < 8; ++r) { rowmax[r] = -1e30f; rowsum[r] = 0.f; }

  const int hi = lane >> 4, col = lane & 15;
  const float scale = 0.125f;               // 1/sqrt(DV=64)

  const int NZT = SEQ / 64;                 // 32 KV tiles
  stageKV(0, 0);
  for (int it = 0; it < NZT; ++it) {
    const int buf = it & 1;
    if (it + 1 < NZT) { stageKV(buf ^ 1, (it + 1) * 64); wait_async4(); }
    else              { wait_async0(); }
    __syncthreads();

    // ---- scores: S = Q K^T  (16 rows x 64 z per wave), f32 accumulate
    v16bf kb[8];
#pragma unroll
    for (int nt = 0; nt < 4; ++nt) {
      kb[2 * nt + 0] = load_frag_b(&Ks[buf][(nt * 16) * 72 + 0],  72, lane);
      kb[2 * nt + 1] = load_frag_b(&Ks[buf][(nt * 16) * 72 + 32], 72, lane);
    }
    FragC s[4];
    for (int j = 0; j < 4; ++j)
      for (int e = 0; e < 8; ++e) s[j].f[e] = 0.f;
#pragma unroll
    for (int nt = 0; nt < 4; ++nt) {
      s[nt].v = __builtin_amdgcn_wmma_f32_16x16x32_bf16(false, qa0, false, kb[2 * nt + 0], (short)0, s[nt].v, false, false);
      s[nt].v = __builtin_amdgcn_wmma_f32_16x16x32_bf16(false, qa1, false, kb[2 * nt + 1], (short)0, s[nt].v, false, false);
    }

    // ---- online softmax per row (row = r + 8*hi of this wave's 16 rows)
#pragma unroll
    for (int r = 0; r < 8; ++r) {
      float lm = fmaxf(fmaxf(s[0].f[r], s[1].f[r]), fmaxf(s[2].f[r], s[3].f[r])) * scale;
      lm = fmaxf(lm, __shfl_xor(lm, 1));
      lm = fmaxf(lm, __shfl_xor(lm, 2));
      lm = fmaxf(lm, __shfl_xor(lm, 4));
      lm = fmaxf(lm, __shfl_xor(lm, 8));
      float mnew  = fmaxf(rowmax[r], lm);
      float alpha = __expf(rowmax[r] - mnew);
      float ls = 0.f;
      const int prow = (wave * 16 + r + 8 * hi) * 72;
#pragma unroll
      for (int nt = 0; nt < 4; ++nt) {
        float p = __expf(s[nt].f[r] * scale - mnew);
        Ps[prow + nt * 16 + col] = f2bf(p);
        ls += p;
      }
      ls += __shfl_xor(ls, 1);
      ls += __shfl_xor(ls, 2);
      ls += __shfl_xor(ls, 4);
      ls += __shfl_xor(ls, 8);
      rowsum[r] = rowsum[r] * alpha + ls;
      rowmax[r] = mnew;
#pragma unroll
      for (int j = 0; j < 4; ++j) o[j].f[r] *= alpha;
    }

    // ---- O += P V  (A = own P slab re-read from LDS; B = Vt, contiguous K=z)
    v16bf pa0 = load_frag_a(&Ps[(wave * 16) * 72 + 0],  72, lane);
    v16bf pa1 = load_frag_a(&Ps[(wave * 16) * 72 + 32], 72, lane);
    v16bf vb[8];
#pragma unroll
    for (int nt = 0; nt < 4; ++nt) {
      vb[2 * nt + 0] = load_frag_b(&Vs[buf][(nt * 16) * 72 + 0],  72, lane);
      vb[2 * nt + 1] = load_frag_b(&Vs[buf][(nt * 16) * 72 + 32], 72, lane);
    }
#pragma unroll
    for (int nt = 0; nt < 4; ++nt) {
      o[nt].v = __builtin_amdgcn_wmma_f32_16x16x32_bf16(false, pa0, false, vb[2 * nt + 0], (short)0, o[nt].v, false, false);
      o[nt].v = __builtin_amdgcn_wmma_f32_16x16x32_bf16(false, pa1, false, vb[2 * nt + 1], (short)0, o[nt].v, false, false);
    }
    __syncthreads();
  }

  // ---- normalize and emit [B*S, H*D] bf16
  const int b = bh >> 4, h = bh & 15;
#pragma unroll
  for (int nt = 0; nt < 4; ++nt)
#pragma unroll
    for (int r = 0; r < 8; ++r) {
      int srow = q0 + wave * 16 + r + 8 * hi;
      float v = o[nt].f[r] / rowsum[r];
      vw[((size_t)(b * SEQ) + srow) * IND + h * DH + nt * 16 + col] = f2bf(v);
    }
}

// ---------------------------------------------------------------------------
// Host launcher
// ---------------------------------------------------------------------------
extern "C" void kernel_launch(void* const* d_in, const int* in_sizes, int n_in,
                              void* d_out, int out_size, void* d_ws, size_t ws_size,
                              hipStream_t stream) {
  (void)in_sizes; (void)n_in; (void)out_size; (void)ws_size;
  const float* x  = (const float*)d_in[0];
  const float* Wq = (const float*)d_in[1];
  const float* bq = (const float*)d_in[2];
  const float* Wk = (const float*)d_in[3];
  const float* bk = (const float*)d_in[4];
  const float* Wv = (const float*)d_in[5];
  const float* bv = (const float*)d_in[6];
  const float* Wo = (const float*)d_in[7];
  const float* bo = (const float*)d_in[8];
  float* out = (float*)d_out;

  // Workspace layout (bf16 as ushort): ~88 MB total
  unsigned short* ws  = (unsigned short*)d_ws;
  unsigned short* xb  = ws;                                // x        8192x1024
  unsigned short* wqb = xb  + (size_t)MTOT * IND;          // Wq       1024x1024
  unsigned short* wkb = wqb + (size_t)IND * IND;
  unsigned short* wvb = wkb + (size_t)IND * IND;
  unsigned short* wob = wvb + (size_t)IND * IND;
  unsigned short* Qb  = wob + (size_t)IND * IND;           // [B,H,S,D]
  unsigned short* Kb  = Qb  + (size_t)MTOT * IND;          // [B,H,S,D]
  unsigned short* Vtb = Kb  + (size_t)MTOT * IND;          // [B,H,D,S]
  unsigned short* vwb = Vtb + (size_t)MTOT * IND;          // [B*S, H*D]

  cvt_kernel<<<(MTOT * IND) / 1024, 256, 0, stream>>>(x,  xb,  MTOT * IND);
  cvt_kernel<<<(IND * IND) / 1024, 256, 0, stream>>>(Wq, wqb, IND * IND);
  cvt_kernel<<<(IND * IND) / 1024, 256, 0, stream>>>(Wk, wkb, IND * IND);
  cvt_kernel<<<(IND * IND) / 1024, 256, 0, stream>>>(Wv, wvb, IND * IND);
  cvt_kernel<<<(IND * IND) / 1024, 256, 0, stream>>>(Wo, wob, IND * IND);

  dim3 g(IND / 128, MTOT / 128);
  gemm_qkvo<<<g, 256, 0, stream>>>(xb, wqb, bq, Qb,  nullptr, 0);
  gemm_qkvo<<<g, 256, 0, stream>>>(xb, wkb, bk, Kb,  nullptr, 0);
  gemm_qkvo<<<g, 256, 0, stream>>>(xb, wvb, bv, Vtb, nullptr, 1);

  attn_kernel<<<dim3(SEQ / 128, BATCH * NH), 256, 0, stream>>>(Qb, Kb, Vtb, vwb);

  gemm_qkvo<<<g, 256, 0, stream>>>(vwb, wob, bo, nullptr, out, 2);
}
